// RecurrentEncoderDecoder_70446053589187
// MI455X (gfx1250) — compile-verified, matrix operand order checked by reading, not run on hardware
//
#include <hip/hip_runtime.h>

// ---------------------------------------------------------------------------
// CDNA5 (gfx1250) bf16 WMMA seq2seq LSTM encoder/decoder
// ---------------------------------------------------------------------------

typedef __attribute__((ext_vector_type(16))) __bf16 v16bf;
typedef __attribute__((ext_vector_type(8)))  float  v8f;

#define WMMA_BF16(A, B, C) \
  __builtin_amdgcn_wmma_f32_16x16x32_bf16(false, (A), false, (B), (short)0, (C), false, false)

static constexpr int BATCH = 64;    // B
static constexpr int HID   = 512;   // H
static constexpr int GATES = 2048;  // 4H
static constexpr int TENC  = 256;
static constexpr int TDEC  = 96;

// gfx1250 native transcendental tanh (TRANS32 op).  The trailing s_delay_alu
// covers the 1-op TRANS result-use hazard (CDNA5 ISA 5.x / S_DELAY_ALU).
__device__ __forceinline__ float tanh_fast(float x) {
#if defined(__gfx1250__)
  float y;
  asm volatile("v_tanh_f32 %0, %1\n\t"
               "s_delay_alu instid0(TRANS32_DEP_1)"
               : "=v"(y) : "v"(x));
  return y;
#else
  return tanhf(x);
#endif
}

// branch-free sigmoid via native tanh: sigma(x) = 0.5*tanh(x/2) + 0.5
__device__ __forceinline__ float sigmoid_f(float x) {
  return __builtin_fmaf(0.5f, tanh_fast(0.5f * x), 0.5f);
}

// ISA 7.12.2 16-bit fragment K mapping:
// lanes 0-15:  e=0..7 -> k=0..7,  e=8..15 -> k=16..23
// lanes 16-31: e=0..7 -> k=8..15, e=8..15 -> k=24..31
__device__ __forceinline__ int frag_k_map(int lane, int e) {
  return (e & 7) + ((e >> 3) << 4) + ((lane >> 4) << 3);
}

// ---------------------------------------------------------------------------
// Convert fp32 weight (2048 x K row-major) into bf16 B-fragment layout grouped
// per consumer wave:  out[jt(32)][kk(K/32)][g(4)][lane(32)][e(16)]
//   n = g*512 + jt*16 + (lane&15),  k = kk*32 + frag_k_map(lane, e)
// ---------------------------------------------------------------------------
__global__ void convert_frags(const float* __restrict__ W,
                              __bf16* __restrict__ out, int K) {
  int idx = blockIdx.x * blockDim.x + threadIdx.x;
  int total = GATES * K;
  if (idx >= total) return;
  int e    = idx & 15;
  int lane = (idx >> 4) & 31;
  int g    = (idx >> 9) & 3;
  int KK   = K >> 5;
  int tmp  = idx >> 11;
  int kk   = tmp % KK;
  int jt   = tmp / KK;
  int n = g * HID + jt * 16 + (lane & 15);
  int k = kk * 32 + frag_k_map(lane, e);
  out[idx] = (__bf16)W[n * K + k];
}

// ---------------------------------------------------------------------------
// xw0: encoder layer-0 input projection (K=9), fully parallel.
// out[dir][t][b][col] = bias[col] + sum_d X[b][t][d] * wih[col][d]
// ---------------------------------------------------------------------------
__global__ void xw0_kernel(const float* __restrict__ X,
                           const float* __restrict__ wih0, const float* __restrict__ wih1,
                           const float* __restrict__ b0,   const float* __restrict__ b1,
                           float* __restrict__ out) {
  unsigned idx = blockIdx.x * blockDim.x + threadIdx.x;
  if (idx >= (unsigned)(2 * TENC * BATCH * 512)) return;
  int col4 = idx & 511;
  int b    = (idx >> 9) & 63;
  int t    = (idx >> 15) & 255;
  int dir  = idx >> 23;
  const float* wih  = dir ? wih1 : wih0;
  const float* bias = dir ? b1 : b0;
  float x[9];
#pragma unroll
  for (int d = 0; d < 9; ++d) x[d] = X[(size_t)(b * TENC + t) * 9 + d];
  float4 r;
  float* rp = &r.x;
#pragma unroll
  for (int c = 0; c < 4; ++c) {
    int col = col4 * 4 + c;
    float a = bias[col];
#pragma unroll
    for (int d = 0; d < 9; ++d) a += x[d] * wih[col * 9 + d];
    rp[c] = a;
  }
  *(float4*)(out + ((size_t)idx << 2)) = r;
}

// ---------------------------------------------------------------------------
// xwdec: static part of decoder layer-0 input projection:
// out[t][b][col] = bias0[col] + sum_{d<8} FF[b][t][d] * wih0[col][d+1]
// (the dynamic y_prev * wih0[col][0] term is added in the decoder)
// ---------------------------------------------------------------------------
__global__ void xwdec_kernel(const float* __restrict__ FF,
                             const float* __restrict__ wih,   // (2048,9)
                             const float* __restrict__ bias,
                             float* __restrict__ out) {
  unsigned idx = blockIdx.x * blockDim.x + threadIdx.x;
  int col4 = idx & 511;
  int b    = (idx >> 9) & 63;
  int t    = idx >> 15;
  if (t >= TDEC) return;
  float x[8];
#pragma unroll
  for (int d = 0; d < 8; ++d) x[d] = FF[(size_t)(b * TDEC + t) * 8 + d];
  float4 r;
  float* rp = &r.x;
#pragma unroll
  for (int c = 0; c < 4; ++c) {
    int col = col4 * 4 + c;
    float a = bias[col];
#pragma unroll
    for (int d = 0; d < 8; ++d) a += x[d] * wih[col * 9 + 1 + d];
    rp[c] = a;
  }
  *(float4*)(out + ((size_t)idx << 2)) = r;
}

// ---------------------------------------------------------------------------
// Unified encoder recurrence (either layer): one workgroup per direction,
// 1024 threads = 32 waves, wave w owns hidden tile column j_tile=w for all 4
// batch tiles.  Accumulators init from precomputed bias-folded xw buffer;
// h in LDS A-fragment layout (double buffered); c in VGPRs.
// ---------------------------------------------------------------------------
__global__ __launch_bounds__(1024) void enc_step_kernel(
    const __bf16* __restrict__ whh0f, const __bf16* __restrict__ whh1f,
    const float* __restrict__ xw,     // [2][256][64][2048], bias folded
    __bf16* __restrict__ seqbuf,      // [256][4][32][32][16]
    int write_seq, int fin_slot,
    float* __restrict__ hfin, float* __restrict__ cfin)
{
  const int dir = blockIdx.x;
  const __bf16* whh = dir ? whh1f : whh0f;
  const float* xwd = xw + (size_t)dir * 16384 * GATES;
  float* hf = hfin + (size_t)(fin_slot + dir) * BATCH * HID;
  float* cf = cfin + (size_t)(fin_slot + dir) * BATCH * HID;

  __shared__ __align__(32) __bf16 hfrag[2][4][16][32][16]; // 128 KB

  const int tid  = threadIdx.x;
  const int lane = tid & 31;
  const int jt   = tid >> 5;
  const int ncol = lane & 15;
  const int lhi  = lane >> 4;

  const __bf16* __restrict__ wl = whh + (size_t)jt * (16 * 4 * 512) + lane * 16;

  for (int i = tid; i < 4 * 16 * 32 * 16; i += 1024)
    (&hfrag[0][0][0][0][0])[i] = (__bf16)0.0f;

  v8f creg[4];
  for (int mt = 0; mt < 4; ++mt)
#pragma unroll
    for (int r = 0; r < 8; ++r) creg[mt][r] = 0.0f;

  for (int s = 0; s < TENC; ++s) {
    const int t   = dir ? (TENC - 1 - s) : s;
    const int cur = s & 1, nxt = cur ^ 1;
    __syncthreads();

    for (int mt = 0; mt < 4; ++mt) {
      v8f acc[4];
#pragma unroll
      for (int g = 0; g < 4; ++g) {
        int col = g * HID + jt * 16 + ncol;
#pragma unroll
        for (int r = 0; r < 8; ++r) {
          int m = mt * 16 + r + lhi * 8;
          acc[g][r] = xwd[(size_t)(t * BATCH + m) * GATES + col];
        }
      }
#pragma unroll 1
      for (int kk = 0; kk < 16; ++kk) {
        v16bf af = *(const v16bf*)&hfrag[cur][mt][kk][lane][0];
#pragma unroll
        for (int g = 0; g < 4; ++g) {
          v16bf bf = *(const v16bf*)(wl + ((kk * 4 + g) << 9));
          acc[g] = WMMA_BF16(af, bf, acc[g]);
        }
      }
      v8f cc = creg[mt];
#pragma unroll
      for (int r = 0; r < 8; ++r) {
        float iv = sigmoid_f(acc[0][r]);
        float fv = sigmoid_f(acc[1][r]);
        float gv = tanh_fast(acc[2][r]);
        float ov = sigmoid_f(acc[3][r]);
        float cn = fv * cc[r] + iv * gv;
        float hn = ov * tanh_fast(cn);
        cc[r] = cn;
        int m = mt * 16 + r + lhi * 8;
        int k = jt * 16 + ncol;
        {
          int kk2 = k >> 5, kr = k & 31;
          int l2 = (m & 15) | (((kr >> 3) & 1) << 4);
          int e2 = (kr & 7) | ((kr >> 4) << 3);
          hfrag[nxt][mt][kk2][l2][e2] = (__bf16)hn;
        }
        if (write_seq) {
          int kg = dir * HID + k;
          int kkg = kg >> 5, krg = kg & 31;
          int lg = (m & 15) | (((krg >> 3) & 1) << 4);
          int eg = (krg & 7) | ((krg >> 4) << 3);
          seqbuf[((((size_t)t * 4 + mt) * 32 + kkg) * 32 + lg) * 16 + eg] = (__bf16)hn;
        }
        if (s == TENC - 1) {
          hf[(size_t)m * HID + k] = hn;
          cf[(size_t)m * HID + k] = cn;
        }
      }
      creg[mt] = cc;
    }
  }
}

// ---------------------------------------------------------------------------
// Full-grid WMMA GEMM: xw1[dir][t*64+b][col] = seq @ W_ih_l1(dir)^T + bias.
// M=16384, N=2048, K=1024.  One wave per (dir, m_tile, j) strip.
// ---------------------------------------------------------------------------
__global__ __launch_bounds__(256) void xw1_gemm_kernel(
    const __bf16* __restrict__ seqbuf,    // [1024 m_tiles][32 kk][32][16]
    const __bf16* __restrict__ wf0, const __bf16* __restrict__ wf1,
    const float* __restrict__ bias0, const float* __restrict__ bias1,
    float* __restrict__ xw)               // [2][16384][2048]
{
  int wid  = (blockIdx.x * blockDim.x + threadIdx.x) >> 5;
  int lane = threadIdx.x & 31;
  int jt    = wid & 31;
  int mtile = (wid >> 5) & 1023;
  int dir   = wid >> 15;
  const __bf16* wf   = dir ? wf1 : wf0;
  const float*  bias = dir ? bias1 : bias0;
  const int ncol = lane & 15, lhi = lane >> 4;

  v8f acc[4];
#pragma unroll
  for (int g = 0; g < 4; ++g) {
    float bv = bias[g * HID + jt * 16 + ncol];
#pragma unroll
    for (int r = 0; r < 8; ++r) acc[g][r] = bv;
  }
  const __bf16* __restrict__ al = seqbuf + (size_t)mtile * (32 * 512) + lane * 16;
  const __bf16* __restrict__ wl = wf + (size_t)jt * (32 * 4 * 512) + lane * 16;
#pragma unroll 4
  for (int kk = 0; kk < 32; ++kk) {
    v16bf a = *(const v16bf*)(al + (kk << 9));
#pragma unroll
    for (int g = 0; g < 4; ++g) {
      v16bf b = *(const v16bf*)(wl + ((kk * 4 + g) << 9));
      acc[g] = WMMA_BF16(a, b, acc[g]);
    }
  }
  float* out = xw + (size_t)dir * 16384 * GATES;
#pragma unroll
  for (int g = 0; g < 4; ++g) {
    int col = g * HID + jt * 16 + ncol;
#pragma unroll
    for (int r = 0; r < 8; ++r) {
      int m = mtile * 16 + r + lhi * 8;
      out[(size_t)m * GATES + col] = acc[g][r];
    }
  }
}

// ---------------------------------------------------------------------------
// Bidirectional state merge: out = elu(s @ bidi_w^T + bidi_b) for both h & c.
// ---------------------------------------------------------------------------
__global__ void merge_kernel(const float* __restrict__ hfin,
                             const float* __restrict__ cfin,
                             const float* __restrict__ Wb,  // (1024,2048)
                             const float* __restrict__ bb,  // (1024)
                             float* __restrict__ hdec,      // [2][64][512]
                             float* __restrict__ cdec)
{
  int idx = blockIdx.x * blockDim.x + threadIdx.x;
  if (idx >= 2 * BATCH * 1024) return;
  int which = idx >> 16;            // 0=h, 1=c
  int col = idx & 1023;
  int b   = (idx >> 10) & 63;
  const float* S = which ? cfin : hfin; // [4][64][512]
  float a = bb[col];
  const float* wr = Wb + (size_t)col * GATES;
  for (int g = 0; g < GATES; ++g)
    a += S[(size_t)(g >> 9) * (BATCH * HID) + b * HID + (g & 511)] * wr[g];
  a = (a > 0.0f) ? a : (__expf(a) - 1.0f);
  float* O = which ? cdec : hdec;
  O[(size_t)(col >> 9) * (BATCH * HID) + b * HID + (col & 511)] = a;
}

// ---------------------------------------------------------------------------
// Decoder: single workgroup, 96 steps, 2 LSTM layers per step.  h0/h1 in
// fragment-layout LDS (double buffered); c0/c1 live in the cdec buffer and
// are updated in place (keeps VGPR pressure low at 1024 threads).
// ---------------------------------------------------------------------------
__global__ __launch_bounds__(1024) void decoder_kernel(
    const float* __restrict__ X,       // (64,256,9) for y0
    const float* __restrict__ xwd0,    // [96][64][2048] static part of layer0
    const __bf16* __restrict__ whh0f, const __bf16* __restrict__ whh1f,
    const __bf16* __restrict__ wih1f,
    const float* __restrict__ wih0,    // (2048,9): only column 0 used here
    const float* __restrict__ bias1,
    const float* __restrict__ outw,    // (9,512)
    const float* __restrict__ outb,    // (9)
    const float* __restrict__ hdec,    // [2][64][512]
    float* __restrict__ cstate,        // [2][64][512], updated in place
    float* __restrict__ Y)             // (64,96,9)
{
  __shared__ __align__(32) __bf16 h0f[2][4][16][32][16]; // 128 KB
  __shared__ __align__(32) __bf16 h1f[2][4][16][32][16]; // 128 KB
  __shared__ float yprev[BATCH];

  const int tid  = threadIdx.x;
  const int lane = tid & 31;
  const int jt   = tid >> 5;
  const int ncol = lane & 15;
  const int lhi  = lane >> 4;

  const __bf16* __restrict__ w0l = whh0f + (size_t)jt * (16 * 4 * 512) + lane * 16;
  const __bf16* __restrict__ wxl = wih1f + (size_t)jt * (16 * 4 * 512) + lane * 16;
  const __bf16* __restrict__ whl = whh1f + (size_t)jt * (16 * 4 * 512) + lane * 16;

  // init h fragments (parity 0) from merged states (fp32 -> bf16)
  for (int i = tid; i < 4 * 16 * 32 * 16; i += 1024) {
    int e = i & 15, ln = (i >> 4) & 31, kk = (i >> 9) & 15, mt = i >> 13;
    int m = mt * 16 + (ln & 15);
    int k = kk * 32 + frag_k_map(ln, e);
    h0f[0][mt][kk][ln][e] = (__bf16)hdec[(size_t)m * HID + k];
    h1f[0][mt][kk][ln][e] = (__bf16)hdec[(size_t)BATCH * HID + m * HID + k];
  }
  if (tid < BATCH) yprev[tid] = X[(size_t)(tid * TENC + 255) * 9 + 0];

  float w0col[4], b1reg[4];
#pragma unroll
  for (int g = 0; g < 4; ++g) {
    int col = g * HID + jt * 16 + ncol;
    w0col[g] = wih0[col * 9 + 0];
    b1reg[g] = bias1[col];
  }
  const int jcol = jt * 16 + ncol;
  __syncthreads();

  for (int t = 0; t < TDEC; ++t) {
    const int cur = t & 1, nxt = cur ^ 1;

    // ---- layer 0 ----
    for (int mt = 0; mt < 4; ++mt) {
      v8f acc[4];
#pragma unroll
      for (int g = 0; g < 4; ++g) {
        int col = g * HID + jcol;
#pragma unroll
        for (int r = 0; r < 8; ++r) {
          int m = mt * 16 + r + lhi * 8;
          acc[g][r] = xwd0[(size_t)(t * BATCH + m) * GATES + col] + yprev[m] * w0col[g];
        }
      }
#pragma unroll 1
      for (int kk = 0; kk < 16; ++kk) {
        v16bf af = *(const v16bf*)&h0f[cur][mt][kk][lane][0];
#pragma unroll
        for (int g = 0; g < 4; ++g) {
          v16bf bf = *(const v16bf*)(w0l + ((kk * 4 + g) << 9));
          acc[g] = WMMA_BF16(af, bf, acc[g]);
        }
      }
#pragma unroll
      for (int r = 0; r < 8; ++r) {
        int m = mt * 16 + r + lhi * 8;
        float co = cstate[(size_t)m * HID + jcol];
        float iv = sigmoid_f(acc[0][r]);
        float fv = sigmoid_f(acc[1][r]);
        float gv = tanh_fast(acc[2][r]);
        float ov = sigmoid_f(acc[3][r]);
        float cn = fv * co + iv * gv;
        float hn = ov * tanh_fast(cn);
        cstate[(size_t)m * HID + jcol] = cn;
        int kk2 = jcol >> 5, kr = jcol & 31;
        int l2 = (m & 15) | (((kr >> 3) & 1) << 4);
        int e2 = (kr & 7) | ((kr >> 4) << 3);
        h0f[nxt][mt][kk2][l2][e2] = (__bf16)hn;
      }
    }
    __syncthreads();

    // ---- layer 1 (x = new h0, recurrent h1) ----
    for (int mt = 0; mt < 4; ++mt) {
      v8f acc[4];
#pragma unroll
      for (int g = 0; g < 4; ++g)
#pragma unroll
        for (int r = 0; r < 8; ++r) acc[g][r] = b1reg[g];
#pragma unroll 1
      for (int kk = 0; kk < 16; ++kk) {
        v16bf ax = *(const v16bf*)&h0f[nxt][mt][kk][lane][0];
        v16bf ah = *(const v16bf*)&h1f[cur][mt][kk][lane][0];
#pragma unroll
        for (int g = 0; g < 4; ++g) {
          v16bf bx = *(const v16bf*)(wxl + ((kk * 4 + g) << 9));
          acc[g] = WMMA_BF16(ax, bx, acc[g]);
          v16bf bh = *(const v16bf*)(whl + ((kk * 4 + g) << 9));
          acc[g] = WMMA_BF16(ah, bh, acc[g]);
        }
      }
#pragma unroll
      for (int r = 0; r < 8; ++r) {
        int m = mt * 16 + r + lhi * 8;
        float co = cstate[(size_t)BATCH * HID + m * HID + jcol];
        float iv = sigmoid_f(acc[0][r]);
        float fv = sigmoid_f(acc[1][r]);
        float gv = tanh_fast(acc[2][r]);
        float ov = sigmoid_f(acc[3][r]);
        float cn = fv * co + iv * gv;
        float hn = ov * tanh_fast(cn);
        cstate[(size_t)BATCH * HID + m * HID + jcol] = cn;
        int kk2 = jcol >> 5, kr = jcol & 31;
        int l2 = (m & 15) | (((kr >> 3) & 1) << 4);
        int e2 = (kr & 7) | ((kr >> 4) << 3);
        h1f[nxt][mt][kk2][l2][e2] = (__bf16)hn;
      }
    }
    __syncthreads();

    // ---- output projection (64x9, K=512) ----
    if (tid < BATCH * 9) {
      int b = tid / 9, q = tid % 9;
      float a = outb[q];
      int mt = b >> 4;
      for (int k = 0; k < HID; ++k) {
        int kk = k >> 5, kr = k & 31;
        int l2 = (b & 15) | (((kr >> 3) & 1) << 4);
        int e2 = (kr & 7) | ((kr >> 4) << 3);
        a += (float)h1f[nxt][mt][kk][l2][e2] * outw[q * HID + k];
      }
      Y[(size_t)(b * TDEC + t) * 9 + q] = a;
      if (q == 0) yprev[b] = a;
    }
    __syncthreads();
  }
}

// ---------------------------------------------------------------------------
// Host launch
// ---------------------------------------------------------------------------
extern "C" void kernel_launch(void* const* d_in, const int* in_sizes, int n_in,
                              void* d_out, int out_size, void* d_ws, size_t ws_size,
                              hipStream_t stream) {
  (void)in_sizes; (void)n_in; (void)out_size; (void)ws_size;
  const float* X            = (const float*)d_in[0];
  const float* FF           = (const float*)d_in[1];
  const float* enc_w_ih_l0  = (const float*)d_in[3];
  const float* enc_w_hh_l0  = (const float*)d_in[4];
  const float* enc_b_l0     = (const float*)d_in[5];
  const float* enc_w_ih_l0r = (const float*)d_in[6];
  const float* enc_w_hh_l0r = (const float*)d_in[7];
  const float* enc_b_l0r    = (const float*)d_in[8];
  const float* enc_w_ih_l1  = (const float*)d_in[9];
  const float* enc_w_hh_l1  = (const float*)d_in[10];
  const float* enc_b_l1     = (const float*)d_in[11];
  const float* enc_w_ih_l1r = (const float*)d_in[12];
  const float* enc_w_hh_l1r = (const float*)d_in[13];
  const float* enc_b_l1r    = (const float*)d_in[14];
  const float* dec_w_ih_l0  = (const float*)d_in[15];
  const float* dec_w_hh_l0  = (const float*)d_in[16];
  const float* dec_b_l0     = (const float*)d_in[17];
  const float* dec_w_ih_l1  = (const float*)d_in[18];
  const float* dec_w_hh_l1  = (const float*)d_in[19];
  const float* dec_b_l1     = (const float*)d_in[20];
  const float* bidi_w_h     = (const float*)d_in[21];
  const float* bidi_b_h     = (const float*)d_in[22];
  const float* out_w        = (const float*)d_in[23];
  const float* out_b        = (const float*)d_in[24];

  size_t off = 0;
  auto alloc = [&](size_t bytes) -> char* {
    char* p = (char*)d_ws + off;
    off += (bytes + 255) & ~(size_t)255;
    return p;
  };

  // fragment-converted bf16 weights (K=512)
  const float* src512[7] = {enc_w_hh_l0, enc_w_hh_l0r, enc_w_hh_l1, enc_w_hh_l1r,
                            dec_w_hh_l0, dec_w_hh_l1, dec_w_ih_l1};
  __bf16* f512[7];
  for (int i = 0; i < 7; ++i) f512[i] = (__bf16*)alloc((size_t)2048 * 512 * 2);
  __bf16* f_wih_l1  = (__bf16*)alloc((size_t)2048 * 1024 * 2);
  __bf16* f_wih_l1r = (__bf16*)alloc((size_t)2048 * 1024 * 2);
  __bf16* seqbuf = (__bf16*)alloc((size_t)256 * 4 * 32 * 32 * 16 * 2); // 33.5 MB
  float*  xw     = (float*)alloc((size_t)2 * 256 * 64 * 2048 * 4);     // 268 MB (xw0 then xw1)
  float*  xwd0   = (float*)alloc((size_t)96 * 64 * 2048 * 4);          // 50 MB
  float*  hfin   = (float*)alloc((size_t)4 * 64 * 512 * 4);
  float*  cfin   = (float*)alloc((size_t)4 * 64 * 512 * 4);
  float*  hdec   = (float*)alloc((size_t)2 * 64 * 512 * 4);
  float*  cdec   = (float*)alloc((size_t)2 * 64 * 512 * 4);

  for (int i = 0; i < 7; ++i)
    convert_frags<<<(2048 * 512) / 256, 256, 0, stream>>>(src512[i], f512[i], 512);
  convert_frags<<<(2048 * 1024) / 256, 256, 0, stream>>>(enc_w_ih_l1, f_wih_l1, 1024);
  convert_frags<<<(2048 * 1024) / 256, 256, 0, stream>>>(enc_w_ih_l1r, f_wih_l1r, 1024);

  // parallel input projections
  xw0_kernel<<<(2 * 256 * 64 * 512) / 256, 256, 0, stream>>>(
      X, enc_w_ih_l0, enc_w_ih_l0r, enc_b_l0, enc_b_l0r, xw);
  xwdec_kernel<<<(96 * 64 * 512 + 255) / 256, 256, 0, stream>>>(
      FF, dec_w_ih_l0, dec_b_l0, xwd0);

  // encoder layer 0 (xw buffer holds xw0)
  enc_step_kernel<<<2, 1024, 0, stream>>>(f512[0], f512[1], xw, seqbuf,
                                          /*write_seq=*/1, /*fin_slot=*/0,
                                          hfin, cfin);
  // layer-1 input projection GEMM overwrites xw (xw0 no longer needed)
  xw1_gemm_kernel<<<8192, 256, 0, stream>>>(seqbuf, f_wih_l1, f_wih_l1r,
                                            enc_b_l1, enc_b_l1r, xw);
  // encoder layer 1
  enc_step_kernel<<<2, 1024, 0, stream>>>(f512[2], f512[3], xw, seqbuf,
                                          /*write_seq=*/0, /*fin_slot=*/2,
                                          hfin, cfin);
  merge_kernel<<<(2 * 64 * 1024 + 255) / 256, 256, 0, stream>>>(
      hfin, cfin, bidi_w_h, bidi_b_h, hdec, cdec);
  decoder_kernel<<<1, 1024, 0, stream>>>(X, xwd0, f512[4], f512[5], f512[6],
                                         dec_w_ih_l0, dec_b_l1,
                                         out_w, out_b, hdec, cdec, (float*)d_out);
}
